// GAT_63007170232683
// MI455X (gfx1250) — compile-verified, matrix operand order checked by reading, not run on hardware
//
#include <hip/hip_runtime.h>
#include <hip/hip_bf16.h>
#include <math.h>

// ---------------------------------------------------------------------------
// Types for CDNA5 WMMA
// ---------------------------------------------------------------------------
typedef __attribute__((ext_vector_type(16))) __bf16          v16bf;
typedef __attribute__((ext_vector_type(8)))  float           v8f;
typedef __attribute__((ext_vector_type(8)))  unsigned short  u16x8;  // 16B

#define NEG_SLOPE 0.2f

// round-to-nearest-even f32 -> bf16 (as raw u16), avoids __bf16 scalar ops
__device__ __forceinline__ unsigned int f32_to_bf16_bits(float f) {
  unsigned int u = __float_as_uint(f);
  unsigned int r = u + 0x7fffu + ((u >> 16) & 1u);
  return r >> 16;
}
__device__ __forceinline__ unsigned int pack_bf16x2(float lo, float hi) {
  return f32_to_bf16_bits(lo) | (f32_to_bf16_bits(hi) << 16);
}

// order-preserving float <-> int encoding for atomicMax-based segment max
__device__ __forceinline__ int enc_f32(float f) {
  int i = __float_as_int(f);
  return i >= 0 ? i : (i ^ 0x7fffffff);
}
__device__ __forceinline__ float dec_f32(int i) {
  return __int_as_float(i >= 0 ? i : (i ^ 0x7fffffff));
}
#define ENC_NEG_INF 0x807fffff  // enc(-inf)

// ---------------------------------------------------------------------------
// init fills
// ---------------------------------------------------------------------------
__global__ void fill_f32_k(float* __restrict__ p, float v, unsigned int n) {
  unsigned int t = blockIdx.x * blockDim.x + threadIdx.x;
  if (t < n) p[t] = v;
}
__global__ void fill_i32_k(int* __restrict__ p, int v, unsigned int n) {
  unsigned int t = blockIdx.x * blockDim.x + threadIdx.x;
  if (t < n) p[t] = v;
}

// ---------------------------------------------------------------------------
// GEMM  C[M x NCOLS] = A[M x K] * B[K x NCOLS]   (f32 in, bf16 WMMA, f32 out)
// Block = 128 threads = 4 waves; each wave owns a 16-row strip x NCOLS.
// K chunked by 32 (one v_wmma_f32_16x16x32_bf16 per 16x16 tile per chunk).
//
// Fragment-friendly LDS layout:
//   sA [row][K]  row-major   -> lane's A frag = 2 contiguous 16B runs
//   sBt[col][K]  transposed  -> lane's B frag = 2 contiguous 16B runs
// Rows padded to 40 halves (80B = 5*16B): bank spread + 16B alignment kept.
// ---------------------------------------------------------------------------
template <int NCOLS>
__global__ void __launch_bounds__(128)
gemm_bf16_wmma_k(const float* __restrict__ A, const float* __restrict__ B,
                 float* __restrict__ C, int M, int K) {
  constexpr int KC  = 32;
  constexpr int NT  = NCOLS / 16;
  constexpr int LDP = 40;  // padded row length in halves (80B, 16B-aligned)
  __shared__ __attribute__((aligned(16))) unsigned short sA[64][LDP];
  __shared__ __attribute__((aligned(16))) unsigned short sBt[NCOLS][LDP];

  const int tid  = threadIdx.x;
  const int wave = tid >> 5;
  const int lane = tid & 31;
  const int rowBase = blockIdx.x * 64;

  v8f acc[NT];
#pragma unroll
  for (int i = 0; i < NT; ++i)
#pragma unroll
    for (int j = 0; j < 8; ++j) acc[i][j] = 0.0f;

  for (int k0 = 0; k0 < K; k0 += KC) {
    // --- stage A tile: 64 x 32, float4 per slot, pack -> one ds_store_b64 ---
    for (int i = tid; i < 64 * (KC / 4); i += 128) {
      int r  = i >> 3;            // / (KC/4)
      int c4 = (i & 7) << 2;      // % (KC/4) * 4
      int gr = rowBase + r;
      float4 v = make_float4(0.f, 0.f, 0.f, 0.f);
      if (gr < M) {
        const float* ap = A + (size_t)gr * K + k0 + c4;
        v = *(const float4*)ap;
        if (k0 + KC < K) __builtin_prefetch(ap + KC, 0, 1);  // global_prefetch_b8
      }
      uint2 pk;
      pk.x = pack_bf16x2(v.x, v.y);
      pk.y = pack_bf16x2(v.z, v.w);
      *(uint2*)&sA[r][c4] = pk;
    }
    // --- stage B tile transposed: thread owns (col, k-quad); coalesced reads ---
    for (int i = tid; i < NCOLS * (KC / 4); i += 128) {
      int c  = i % NCOLS;
      int kq = (i / NCOLS) << 2;
      const float* bp = B + (size_t)(k0 + kq) * NCOLS + c;
      float b0 = bp[0];
      float b1 = bp[NCOLS];
      float b2 = bp[2 * NCOLS];
      float b3 = bp[3 * NCOLS];
      uint2 pk;
      pk.x = pack_bf16x2(b0, b1);
      pk.y = pack_bf16x2(b2, b3);
      *(uint2*)&sBt[c][kq] = pk;
    }
    __syncthreads();

    // A fragment: lane l<16 -> row m, K {0..7,16..23}; lane>=16 -> K {8..15,24..31}
    const int m  = wave * 16 + (lane & 15);
    const int kb = (lane >> 4) * 8;
    union { v16bf v; u16x8 h[2]; } af;
    {
      const unsigned short* ar = &sA[m][0];
      af.h[0] = *(const u16x8*)(ar + kb);        // ds_load_b128
      af.h[1] = *(const u16x8*)(ar + 16 + kb);   // ds_load_b128
    }

#pragma unroll
    for (int nt = 0; nt < NT; ++nt) {
      const int col = nt * 16 + (lane & 15);
      union { v16bf v; u16x8 h[2]; } bf;
      const unsigned short* br = &sBt[col][0];
      bf.h[0] = *(const u16x8*)(br + kb);        // ds_load_b128
      bf.h[1] = *(const u16x8*)(br + 16 + kb);   // ds_load_b128
      acc[nt] = __builtin_amdgcn_wmma_f32_16x16x32_bf16(
          false, af.v, false, bf.v, (short)0, acc[nt], false, false);
    }
    __syncthreads();
  }

  // C layout: VGPR v, lanes 0-15 -> M=v, lanes 16-31 -> M=v+8; N = lane&15
  const int rs = rowBase + wave * 16 + (lane >> 4) * 8;
  const int cn = lane & 15;
#pragma unroll
  for (int nt = 0; nt < NT; ++nt)
#pragma unroll
    for (int v = 0; v < 8; ++v) {
      int gr = rs + v;
      if (gr < M) C[(size_t)gr * NCOLS + nt * 16 + cn] = acc[nt][v];
    }
}

// ---------------------------------------------------------------------------
// per-(node,head) attention logits: al = <h[n,h,:], a[h,:]>
// ---------------------------------------------------------------------------
__global__ void attn_logits_k(const float* __restrict__ h,
                              const float* __restrict__ a_src,
                              const float* __restrict__ a_dst,
                              float* __restrict__ als, float* __restrict__ ald,
                              unsigned int Nn, int H, int Cc) {
  unsigned int t = blockIdx.x * blockDim.x + threadIdx.x;
  if (t >= Nn * (unsigned)H) return;
  unsigned int n = t / H;
  int hh = (int)(t % H);
  const float* hp = h + (size_t)n * H * Cc + (size_t)hh * Cc;
  float s = 0.f, d = 0.f;
  for (int c = 0; c < Cc; ++c) {
    float v = hp[c];
    s += v * a_src[hh * Cc + c];
    d += v * a_dst[hh * Cc + c];
  }
  als[t] = s;
  ald[t] = d;
}

// ---------------------------------------------------------------------------
// edge pass 1: e = leakyrelu(al_src[s]+al_dst[d]); segment max via atomicMax
// edges [0,E) from edge_index; edges [E,E+N) are virtual self loops
// ---------------------------------------------------------------------------
__global__ void edge_lrelu_max_k(const int* __restrict__ ei,
                                 const float* __restrict__ als,
                                 const float* __restrict__ ald,
                                 float* __restrict__ ebuf,
                                 int* __restrict__ mEnc,
                                 int E, int Nn, int H) {
  unsigned int t = blockIdx.x * blockDim.x + threadIdx.x;
  unsigned int tot = (unsigned int)(E + Nn) * (unsigned int)H;
  if (t >= tot) return;
  int e  = (int)(t / H);
  int hh = (int)(t % H);
  int s, d;
  if (e < E) { s = ei[e]; d = ei[(size_t)E + e]; } else { s = d = e - E; }
  float v = als[(size_t)s * H + hh] + ald[(size_t)d * H + hh];
  v = v > 0.f ? v : NEG_SLOPE * v;
  ebuf[t] = v;
  atomicMax(mEnc + (size_t)d * H + hh, enc_f32(v));
}

// edge pass 2: ex = exp(e - m[d]); segment sum via atomicAdd
__global__ void edge_exp_sum_k(const int* __restrict__ ei,
                               float* __restrict__ ebuf,
                               const int* __restrict__ mEnc,
                               float* __restrict__ sbuf,
                               int E, int Nn, int H) {
  unsigned int t = blockIdx.x * blockDim.x + threadIdx.x;
  unsigned int tot = (unsigned int)(E + Nn) * (unsigned int)H;
  if (t >= tot) return;
  int e  = (int)(t / H);
  int hh = (int)(t % H);
  int d;
  if (e < E) { d = ei[(size_t)E + e]; } else { d = e - E; }
  float ex = __expf(ebuf[t] - dec_f32(mEnc[(size_t)d * H + hh]));
  ebuf[t] = ex;
  atomicAdd(sbuf + (size_t)d * H + hh, ex);
}

// edge pass 3: out[d,:] += alpha * h[s,:]   (one thread per edge-channel)
__global__ void edge_agg_k(const int* __restrict__ ei,
                           const float* __restrict__ h,
                           const float* __restrict__ ex,
                           const float* __restrict__ sum,
                           float* __restrict__ out,
                           int E, int Nn, int H, int Cc) {
  unsigned int t = blockIdx.x * blockDim.x + threadIdx.x;
  const int F = H * Cc;
  unsigned int tot = (unsigned int)(E + Nn) * (unsigned int)F;
  if (t >= tot) return;
  int e = (int)(t / F);
  int c = (int)(t % F);
  int hh = c / Cc;
  int s, d;
  if (e < E) { s = ei[e]; d = ei[(size_t)E + e]; } else { s = d = e - E; }
  float alpha = ex[(size_t)e * H + hh] / sum[(size_t)d * H + hh];
  atomicAdd(out + (size_t)d * F + c, h[(size_t)s * F + c] * alpha);
}

// bias + ELU in place (conv1 output -> conv2 input)
__global__ void bias_elu_k(float* __restrict__ p, const float* __restrict__ b,
                           unsigned int n, int F) {
  unsigned int t = blockIdx.x * blockDim.x + threadIdx.x;
  if (t >= n) return;
  float v = p[t] + b[t % F];
  p[t] = v > 0.f ? v : expm1f(v);
}

// final: + bias, log_softmax over 16 classes
__global__ void logsoftmax_k(const float* __restrict__ agg,
                             const float* __restrict__ b,
                             float* __restrict__ out, unsigned int Nn) {
  unsigned int n = blockIdx.x * blockDim.x + threadIdx.x;
  if (n >= Nn) return;
  float v[16];
  float mx = -INFINITY;
#pragma unroll
  for (int c = 0; c < 16; ++c) {
    v[c] = agg[(size_t)n * 16 + c] + b[c];
    mx = fmaxf(mx, v[c]);
  }
  float s = 0.f;
#pragma unroll
  for (int c = 0; c < 16; ++c) s += __expf(v[c] - mx);
  float ls = __logf(s);
#pragma unroll
  for (int c = 0; c < 16; ++c) out[(size_t)n * 16 + c] = v[c] - mx - ls;
}

// ---------------------------------------------------------------------------
// launch
// ---------------------------------------------------------------------------
static inline unsigned int cdiv(size_t a, unsigned int b) {
  return (unsigned int)((a + b - 1) / b);
}

extern "C" void kernel_launch(void* const* d_in, const int* in_sizes, int n_in,
                              void* d_out, int out_size, void* d_ws, size_t ws_size,
                              hipStream_t stream) {
  const float* x   = (const float*)d_in[0];
  const int*   ei  = (const int*)d_in[1];
  const float* W1  = (const float*)d_in[2];
  const float* a1s = (const float*)d_in[3];
  const float* a1d = (const float*)d_in[4];
  const float* b1  = (const float*)d_in[5];
  const float* W2  = (const float*)d_in[6];
  const float* a2s = (const float*)d_in[7];
  const float* a2d = (const float*)d_in[8];
  const float* b2  = (const float*)d_in[9];
  float* out = (float*)d_out;

  const int FIN = 512, H1 = 8, C1 = 8, F1 = H1 * C1, C2 = 16;
  const int Nn = in_sizes[0] / FIN;
  const int E  = in_sizes[1] / 2;
  const int EE = E + Nn;

  // workspace carve-out (256B aligned segments)
  char* wp = (char*)d_ws;
  auto alloc = [&](size_t elems) -> void* {
    void* p = (void*)wp;
    wp += ((elems * 4 + 255) & ~(size_t)255);
    return p;
  };
  float* h1   = (float*)alloc((size_t)Nn * F1);   // x @ W1
  float* agg1 = (float*)alloc((size_t)Nn * F1);   // conv1 aggregate -> ELU'd feat
  float* al1s = (float*)alloc((size_t)Nn * H1);
  float* al1d = (float*)alloc((size_t)Nn * H1);
  float* s1   = (float*)alloc((size_t)Nn * H1);
  int*   m1   = (int*)  alloc((size_t)Nn * H1);
  float* e1   = (float*)alloc((size_t)EE * H1);   // per-edge e / alpha-num
  float* g2   = (float*)alloc((size_t)Nn * C2);   // feat @ W2
  float* agg2 = (float*)alloc((size_t)Nn * C2);
  float* al2s = (float*)alloc((size_t)Nn);
  float* al2d = (float*)alloc((size_t)Nn);
  float* s2   = (float*)alloc((size_t)Nn);
  int*   m2   = (int*)  alloc((size_t)Nn);
  float* e2   = (float*)alloc((size_t)EE);

  const unsigned int B = 256;

  // deterministic init of accumulators (graph-replay safe)
  fill_i32_k<<<cdiv((size_t)Nn * H1, B), B, 0, stream>>>(m1, ENC_NEG_INF, Nn * H1);
  fill_f32_k<<<cdiv((size_t)Nn * H1, B), B, 0, stream>>>(s1, 0.f, Nn * H1);
  fill_f32_k<<<cdiv((size_t)Nn * F1, B), B, 0, stream>>>(agg1, 0.f, Nn * F1);
  fill_i32_k<<<cdiv((size_t)Nn, B), B, 0, stream>>>(m2, ENC_NEG_INF, Nn);
  fill_f32_k<<<cdiv((size_t)Nn, B), B, 0, stream>>>(s2, 0.f, Nn);
  fill_f32_k<<<cdiv((size_t)Nn * C2, B), B, 0, stream>>>(agg2, 0.f, Nn * C2);

  // ---------------- conv1 ----------------
  gemm_bf16_wmma_k<64><<<cdiv((size_t)Nn, 64), 128, 0, stream>>>(x, W1, h1, Nn, FIN);
  attn_logits_k<<<cdiv((size_t)Nn * H1, B), B, 0, stream>>>(h1, a1s, a1d, al1s, al1d, Nn, H1, C1);
  edge_lrelu_max_k<<<cdiv((size_t)EE * H1, B), B, 0, stream>>>(ei, al1s, al1d, e1, m1, E, Nn, H1);
  edge_exp_sum_k<<<cdiv((size_t)EE * H1, B), B, 0, stream>>>(ei, e1, m1, s1, E, Nn, H1);
  edge_agg_k<<<cdiv((size_t)EE * F1, B), B, 0, stream>>>(ei, h1, e1, s1, agg1, E, Nn, H1, C1);
  bias_elu_k<<<cdiv((size_t)Nn * F1, B), B, 0, stream>>>(agg1, b1, Nn * F1, F1);

  // ---------------- conv2 ----------------
  gemm_bf16_wmma_k<16><<<cdiv((size_t)Nn, 64), 128, 0, stream>>>(agg1, W2, g2, Nn, F1);
  attn_logits_k<<<cdiv((size_t)Nn, B), B, 0, stream>>>(g2, a2s, a2d, al2s, al2d, Nn, 1, C2);
  edge_lrelu_max_k<<<cdiv((size_t)EE, B), B, 0, stream>>>(ei, al2s, al2d, e2, m2, E, Nn, 1);
  edge_exp_sum_k<<<cdiv((size_t)EE, B), B, 0, stream>>>(ei, e2, m2, s2, E, Nn, 1);
  edge_agg_k<<<cdiv((size_t)EE * C2, B), B, 0, stream>>>(ei, g2, e2, s2, agg2, E, Nn, 1, C2);

  // ---------------- log_softmax ----------------
  logsoftmax_k<<<cdiv((size_t)Nn, B), B, 0, stream>>>(agg2, b2, out, Nn);
}
